// MHA_45268955300511
// MI455X (gfx1250) — compile-verified, matrix operand order checked by reading, not run on hardware
//
#include <hip/hip_runtime.h>
#include <hip/hip_bf16.h>
#include <math.h>

// Problem constants (B,T,C,H) = (4, 2048, 1024, 16), DH = 64
#define BB 4
#define TT 2048
#define CC 1024
#define HH 16
#define DHH 64
#define BT (BB * TT)   // 8192

typedef __attribute__((ext_vector_type(16))) __bf16 v16bf;
typedef __attribute__((ext_vector_type(8)))  __bf16 v8bf;
typedef __attribute__((ext_vector_type(4)))  __bf16 v4bf;
typedef __attribute__((ext_vector_type(8)))  float  v8f;

static __device__ __forceinline__ __bf16 f2bf(float f) {
  // round-to-nearest-even f32 -> bf16
  unsigned u = __builtin_bit_cast(unsigned, f);
  u += 0x7fffu + ((u >> 16) & 1u);
  unsigned short h = (unsigned short)(u >> 16);
  return __builtin_bit_cast(__bf16, h);
}

// Load a 16x32 bf16 WMMA fragment whose 16 "lines" are rows of `base` with
// element stride `stride`, contraction (K) dimension contiguous in memory.
// Per ISA 7.12.2 (16-bit A 16x32): lanes 0-15 hold line l, K={0..7,16..23};
// lanes 16-31 hold line l-16, K={8..15,24..31}. Two 16B loads per lane.
static __device__ __forceinline__ v16bf load_frag_rows(const __bf16* __restrict__ base,
                                                       int stride, int lane) {
  const int line = lane & 15;
  const int ko   = (lane & 16) ? 8 : 0;
  const __bf16* p = base + (size_t)line * (size_t)stride + ko;
  v8bf a = *(const v8bf*)(p);
  v8bf b = *(const v8bf*)(p + 16);
  v16bf r;
#pragma unroll
  for (int i = 0; i < 8; ++i) { r[i] = a[i]; r[i + 8] = b[i]; }
  return r;
}

// ---------------- elementwise f32 -> bf16 (vectorized x4) ----------------
__global__ void cvt_bf16_v4(const float4* __restrict__ in, v4bf* __restrict__ out, int n4) {
  int i = blockIdx.x * blockDim.x + threadIdx.x;
  if (i < n4) {
    float4 f = in[i];
    v4bf o;
    o[0] = f2bf(f.x); o[1] = f2bf(f.y); o[2] = f2bf(f.z); o[3] = f2bf(f.w);
    out[i] = o;
  }
}

// ---------------- GEMM: out[M,N] = sum_k A[m,k] * W[n,k] -------------------
// A: MxK bf16 row-major, W: NxK bf16 row-major (out = A @ W^T), out f32.
// N,K compile-time so stores use immediate offsets and K-loop strength-reduces.
// Block = 256 threads = 8 waves (4 along M x 2 along N); wave tile 64x32.
template <int N, int K>
__global__ __launch_bounds__(256) void gemm_bf16_wmma(const __bf16* __restrict__ A,
                                                      const __bf16* __restrict__ W,
                                                      float* __restrict__ out) {
  const int lane = threadIdx.x & 31;
  const int wave = threadIdx.x >> 5;
  const int m0 = blockIdx.x * 256 + (wave & 3) * 64;
  const int n0 = blockIdx.y * 64 + (wave >> 2) * 32;

  v8f acc[4][2] = {};
  for (int k = 0; k < K; k += 32) {
    v16bf a[4], b[2];
#pragma unroll
    for (int i = 0; i < 4; ++i)
      a[i] = load_frag_rows(A + (size_t)(m0 + 16 * i) * K + k, K, lane);
#pragma unroll
    for (int j = 0; j < 2; ++j)
      b[j] = load_frag_rows(W + (size_t)(n0 + 16 * j) * K + k, K, lane);
#pragma unroll
    for (int i = 0; i < 4; ++i)
#pragma unroll
      for (int j = 0; j < 2; ++j)
        acc[i][j] = __builtin_amdgcn_wmma_f32_16x16x32_bf16(
            false, a[i], false, b[j], (short)0, acc[i][j], false, false);
  }

  const int hhi  = (lane & 16) ? 8 : 0;  // C/D: VGPR r -> M = r + hhi
  const int ncol = lane & 15;            //      N = lane & 15
#pragma unroll
  for (int i = 0; i < 4; ++i)
#pragma unroll
    for (int j = 0; j < 2; ++j) {
      float* p = out + (size_t)(m0 + 16 * i + hhi) * N + (n0 + 16 * j + ncol);
#pragma unroll
      for (int r = 0; r < 8; ++r) p[r * N] = acc[i][j][r];  // imm offsets
    }
}

// -------- RoPE + head split: qkv f32 [B*T,3C] -> q,k [BH,T,DH], vT [BH,DH,T] ---
// Q is pre-scaled by 1/sqrt(DH) = 0.125 (exact power-of-two, folded here).
__global__ void rope_split(const float* __restrict__ qkv, const float* __restrict__ cosT,
                           const float* __restrict__ sinT, __bf16* __restrict__ qo,
                           __bf16* __restrict__ ko, __bf16* __restrict__ vto) {
  int idx = blockIdx.x * blockDim.x + threadIdx.x;  // B*T*H*(DH/2) = 4M
  if (idx >= BB * TT * HH * (DHH / 2)) return;
  const int i = idx & 31;            // rotary pair index
  const int h = (idx >> 5) & (HH - 1);
  const int t = (idx >> 9) & (TT - 1);
  const int b = idx >> 20;

  const float* row = qkv + (size_t)(b * TT + t) * (3 * CC);
  const int d0 = h * DHH + 2 * i;
  const float c = cosT[t * (DHH / 2) + i];
  const float s = sinT[t * (DHH / 2) + i];
  const float q1 = row[d0],          q2 = row[d0 + 1];
  const float k1 = row[CC + d0],     k2 = row[CC + d0 + 1];
  const float v1 = row[2 * CC + d0], v2 = row[2 * CC + d0 + 1];

  const size_t bh = (size_t)b * HH + h;
  const size_t qkoff = (bh * TT + t) * DHH + 2 * i;
  qo[qkoff]     = f2bf((q1 * c - q2 * s) * 0.125f);
  qo[qkoff + 1] = f2bf((q1 * s + q2 * c) * 0.125f);
  ko[qkoff]     = f2bf(k1 * c - k2 * s);
  ko[qkoff + 1] = f2bf(k1 * s + k2 * c);
  vto[(bh * DHH + 2 * i) * TT + t]     = f2bf(v1);
  vto[(bh * DHH + 2 * i + 1) * TT + t] = f2bf(v2);
}

// -------- Flash attention (causal), one wave per 32-query tile ----------------
// q,k: [BH, T, DH] bf16 ; vt: [BH, DH, T] bf16 ; y: [B*T, C] bf16
// Per 32-key step: 8 WMMA for S, 8 WMMA for P@V, 2 WMMA for row-sums (P @ ones):
// D[M,N] = sum_k P[M,k] lands in every column with VGPR r <-> row M — exactly
// the lrow indexing, so the sum shuffle-reduction tree disappears.
__global__ __launch_bounds__(128) void flash_attn(const __bf16* __restrict__ q,
                                                  const __bf16* __restrict__ k,
                                                  const __bf16* __restrict__ vt,
                                                  __bf16* __restrict__ y) {
  __shared__ __bf16 plds[4][32 * 34];  // per-wave 32x32 P tile, padded stride 34

  const int lane  = threadIdx.x & 31;
  const int wave  = threadIdx.x >> 5;
  const int wid   = blockIdx.x * 4 + wave;
  const int bh    = wid >> 6;           // 64 q-tiles (of 32) per (b,h)
  const int qbase = (wid & 63) * 32;
  const int b = bh >> 4, h = bh & 15;
  const int hhi  = (lane & 16) ? 8 : 0;
  const int ncol = lane & 15;

  v16bf onesf;  // all-ones bf16 fragment for row-sum WMMA
#pragma unroll
  for (int i = 0; i < 16; ++i)
    onesf[i] = __builtin_bit_cast(__bf16, (unsigned short)0x3F80);

  v16bf qf[2][2];
#pragma unroll
  for (int mi = 0; mi < 2; ++mi) {
    const __bf16* qp = q + ((size_t)bh * TT + qbase + 16 * mi) * DHH;
    qf[mi][0] = load_frag_rows(qp, DHH, lane);
    qf[mi][1] = load_frag_rows(qp + 32, DHH, lane);
  }

  v8f o[2][4] = {};
  float mrow[2][8], lrow[2][8];
#pragma unroll
  for (int mi = 0; mi < 2; ++mi)
#pragma unroll
    for (int r = 0; r < 8; ++r) { mrow[mi][r] = -INFINITY; lrow[mi][r] = 0.f; }

  __bf16* pw = &plds[wave][0];

  for (int jb = 0; jb <= qbase; jb += 32) {  // 32 keys per step
    v8f s[2][2] = {};
    {
      const __bf16* kp = k + ((size_t)bh * TT + jb) * DHH;
      const v16bf k00 = load_frag_rows(kp, DHH, lane);
      const v16bf k01 = load_frag_rows(kp + 32, DHH, lane);
      const v16bf k10 = load_frag_rows(kp + 16 * DHH, DHH, lane);
      const v16bf k11 = load_frag_rows(kp + 16 * DHH + 32, DHH, lane);
#pragma unroll
      for (int mi = 0; mi < 2; ++mi) {
        s[mi][0] = __builtin_amdgcn_wmma_f32_16x16x32_bf16(false, qf[mi][0], false, k00, (short)0, s[mi][0], false, false);
        s[mi][0] = __builtin_amdgcn_wmma_f32_16x16x32_bf16(false, qf[mi][1], false, k01, (short)0, s[mi][0], false, false);
        s[mi][1] = __builtin_amdgcn_wmma_f32_16x16x32_bf16(false, qf[mi][0], false, k10, (short)0, s[mi][1], false, false);
        s[mi][1] = __builtin_amdgcn_wmma_f32_16x16x32_bf16(false, qf[mi][1], false, k11, (short)0, s[mi][1], false, false);
      }
    }

    float alpha[2][8];
#pragma unroll
    for (int mi = 0; mi < 2; ++mi) {
      const int q0 = qbase + 16 * mi;
      // causal mask (element (M,N): query q0+hhi+r, key jb + sub*16 + ncol)
      if (jb + 15 > q0) {
#pragma unroll
        for (int r = 0; r < 8; ++r)
          if (jb + ncol > q0 + hhi + r) s[mi][0][r] = -INFINITY;
      }
      if (jb + 31 > q0) {
#pragma unroll
        for (int r = 0; r < 8; ++r)
          if (jb + 16 + ncol > q0 + hhi + r) s[mi][1][r] = -INFINITY;
      }

      // running max per row (row lives at VGPR index, replicated in 16-lane half)
#pragma unroll
      for (int r = 0; r < 8; ++r) {
        float v = fmaxf(s[mi][0][r], s[mi][1][r]);
        v = fmaxf(v, __shfl_xor(v, 1, 32));
        v = fmaxf(v, __shfl_xor(v, 2, 32));
        v = fmaxf(v, __shfl_xor(v, 4, 32));
        v = fmaxf(v, __shfl_xor(v, 8, 32));
        const float mn = fmaxf(mrow[mi][r], v);
        alpha[mi][r] = __expf(mrow[mi][r] - mn);
        mrow[mi][r] = mn;
      }
#pragma unroll
      for (int r = 0; r < 8; ++r) {
        s[mi][0][r] = __expf(s[mi][0][r] - mrow[mi][r]);
        s[mi][1][r] = __expf(s[mi][1][r] - mrow[mi][r]);
        lrow[mi][r] *= alpha[mi][r];
      }
#pragma unroll
      for (int c = 0; c < 4; ++c)
#pragma unroll
        for (int r = 0; r < 8; ++r) o[mi][c][r] *= alpha[mi][r];

      // P tile -> LDS (C/D layout out, A-frag layout back in)
#pragma unroll
      for (int r = 0; r < 8; ++r) {
        pw[(16 * mi + hhi + r) * 34 + ncol]      = f2bf(s[mi][0][r]);
        pw[(16 * mi + hhi + r) * 34 + 16 + ncol] = f2bf(s[mi][1][r]);
      }
    }

    v16bf pa[2];
    {
      const int line = lane & 15;
      const int kofs = (lane & 16) ? 8 : 0;
#pragma unroll
      for (int mi = 0; mi < 2; ++mi) {
        const __bf16* pr = pw + (16 * mi + line) * 34 + kofs;
#pragma unroll
        for (int i2 = 0; i2 < 8; ++i2) { pa[mi][i2] = pr[i2]; pa[mi][8 + i2] = pr[16 + i2]; }
      }
    }

    // row sums of P via WMMA with all-ones B: replaces the shuffle tree
#pragma unroll
    for (int mi = 0; mi < 2; ++mi) {
      v8f z = {};
      const v8f lsum = __builtin_amdgcn_wmma_f32_16x16x32_bf16(
          false, pa[mi], false, onesf, (short)0, z, false, false);
#pragma unroll
      for (int r = 0; r < 8; ++r) lrow[mi][r] += lsum[r];
    }

    // O += P @ V  (vT rows = output dims, contiguous along keys; vf reused by both mi)
#pragma unroll
    for (int c = 0; c < 4; ++c) {
      const v16bf vf =
          load_frag_rows(vt + ((size_t)bh * DHH + c * 16) * TT + jb, TT, lane);
#pragma unroll
      for (int mi = 0; mi < 2; ++mi)
        o[mi][c] = __builtin_amdgcn_wmma_f32_16x16x32_bf16(false, pa[mi], false, vf,
                                                           (short)0, o[mi][c], false, false);
    }
  }

  // normalize (one reciprocal per row) and write y[b, t, h*64 + c*16 + ncol]
#pragma unroll
  for (int mi = 0; mi < 2; ++mi)
#pragma unroll
    for (int r = 0; r < 8; ++r) {
      const float inv = 1.0f / lrow[mi][r];
      const int t = qbase + 16 * mi + hhi + r;
      __bf16* yp = y + (size_t)(b * TT + t) * CC + h * DHH + ncol;
#pragma unroll
      for (int c = 0; c < 4; ++c) yp[c * 16] = f2bf(o[mi][c][r] * inv);
    }
}

// ----------------------------------------------------------------------------
extern "C" void kernel_launch(void* const* d_in, const int* in_sizes, int n_in,
                              void* d_out, int out_size, void* d_ws, size_t ws_size,
                              hipStream_t stream) {
  (void)in_sizes; (void)n_in; (void)out_size; (void)ws_size;
  const float* x    = (const float*)d_in[0];
  const float* cosT = (const float*)d_in[1];
  const float* sinT = (const float*)d_in[2];
  const float* Wqkv = (const float*)d_in[3];
  const float* Wo   = (const float*)d_in[4];
  float* out = (float*)d_out;

  char* ws = (char*)d_ws;
  size_t off = 0;
  auto wsalloc = [&](size_t nbytes) -> void* {
    void* p = (void*)(ws + off);
    off += (nbytes + 255) & ~(size_t)255;
    return p;
  };
  __bf16* xbf  = (__bf16*)wsalloc((size_t)BT * CC * 2);          // 16 MB
  __bf16* wqbf = (__bf16*)wsalloc((size_t)3 * CC * CC * 2);      //  6 MB
  __bf16* wobf = (__bf16*)wsalloc((size_t)CC * CC * 2);          //  2 MB
  float*  qkv  = (float*) wsalloc((size_t)BT * 3 * CC * 4);      // 96 MB
  __bf16* qb   = (__bf16*)wsalloc((size_t)BB * HH * TT * DHH * 2);
  __bf16* kb   = (__bf16*)wsalloc((size_t)BB * HH * TT * DHH * 2);
  __bf16* vtb  = (__bf16*)wsalloc((size_t)BB * HH * DHH * TT * 2);
  __bf16* yb   = (__bf16*)wsalloc((size_t)BT * CC * 2);          // 16 MB

  // 1) convert inputs to bf16 (vectorized x4)
  cvt_bf16_v4<<<(BT * CC / 4 + 255) / 256, 256, 0, stream>>>(
      (const float4*)x, (v4bf*)xbf, BT * CC / 4);
  cvt_bf16_v4<<<(3 * CC * CC / 4 + 255) / 256, 256, 0, stream>>>(
      (const float4*)Wqkv, (v4bf*)wqbf, 3 * CC * CC / 4);
  cvt_bf16_v4<<<(CC * CC / 4 + 255) / 256, 256, 0, stream>>>(
      (const float4*)Wo, (v4bf*)wobf, CC * CC / 4);

  // 2) qkv = x @ Wqkv^T  (8192x1024 @ 1024x3072)
  gemm_bf16_wmma<3 * CC, CC><<<dim3(BT / 256, (3 * CC) / 64), 256, 0, stream>>>(
      xbf, wqbf, qkv);

  // 3) RoPE + head split (+ V transpose, Q pre-scaled by 1/8)
  rope_split<<<(BB * TT * HH * (DHH / 2)) / 256, 256, 0, stream>>>(
      qkv, cosT, sinT, qb, kb, vtb);

  // 4) causal flash attention -> y bf16 [B*T, C]  (one wave per 32 queries)
  flash_attn<<<(BB * HH * (TT / 32)) / 4, 128, 0, stream>>>(qb, kb, vtb, yb);

  // 5) out = y @ Wo^T  (f32 result)
  gemm_bf16_wmma<CC, CC><<<dim3(BT / 256, CC / 64), 256, 0, stream>>>(
      yb, wobf, out);
}
